// DisentangledSelfAttention_6193342841114
// MI455X (gfx1250) — matrix operand order repaired and checked by reading, not compiled, not run on hardware
//
#include <hip/hip_runtime.h>
#include <hip/hip_bf16.h>

typedef __attribute__((ext_vector_type(16))) _Float16 v16h;
typedef __attribute__((ext_vector_type(8)))  float    v8f;

#define WMMA_F16(a,b,c) __builtin_amdgcn_wmma_f32_16x16x32_f16(false,(a),false,(b),(short)0,(c),false,false)

// Problem constants
// B=2, S=2048, D=64, H=12, OH=768, 2K=1024

// ---------------------------------------------------------------------------
// Projection kernel: one wave (32 threads) per 16x16 output tile.
// jobs: 0:Q 1:K 2:V(->transposed VT) 3:pk 4:pq.  GEMM is [M,64] @ [768,64]^T.
// A layout (16x32 f16): lane m=L%16; element e -> K = half*16 + e
// B layout (32x16 f16): lane n=L%16; element e -> K = half*16 + e
// C layout (16x16 f32): lane -> n=L%16 ; vgpr r -> m = r + 8*(L/16)
// ---------------------------------------------------------------------------
__global__ __launch_bounds__(32) void proj_kernel(
    const float* __restrict__ H,  const float* __restrict__ P,
    const float* __restrict__ Wq, const float* __restrict__ bq,
    const float* __restrict__ Wk, const float* __restrict__ bk,
    const float* __restrict__ Wv, const float* __restrict__ bv,
    const float* __restrict__ Wpk,const float* __restrict__ bpk,
    const float* __restrict__ Wpq,const float* __restrict__ bpq,
    _Float16* __restrict__ Qh, _Float16* __restrict__ Kh,
    _Float16* __restrict__ VT, _Float16* __restrict__ pkP,
    _Float16* __restrict__ pqP)
{
  const int lane = threadIdx.x, half = lane >> 4, lh = lane & 15;
  int idx = blockIdx.x;
  int job, tm, tn;
  if (idx < 3 * 12288) {            // Q/K/V: 256 row tiles x 48 col tiles
    job = idx / 12288; int r = idx % 12288; tm = r / 48; tn = r % 48;
  } else {                          // pk/pq: 64 row tiles x 48 col tiles
    int r2 = idx - 3 * 12288; job = 3 + r2 / 3072;
    int r = r2 % 3072; tm = r / 48; tn = r % 48;
  }
  const float* X = (job < 3) ? H : P;
  const float* W; const float* bias;
  switch (job) {
    case 0: W = Wq;  bias = bq;  break;
    case 1: W = Wk;  bias = bk;  break;
    case 2: W = Wv;  bias = bv;  break;
    case 3: W = Wpk; bias = bpk; break;
    default:W = Wpq; bias = bpq; break;
  }

  // A tile: 16 rows of X, K = 64 (two 16x32 halves), convert f32 -> f16
  const float* xr = X + (size_t)(tm * 16 + lh) * 64;
  v16h a0, a1;
  #pragma unroll
  for (int e = 0; e < 16; ++e) {
    int k = half * 16 + e;
    a0[e] = (_Float16)xr[k];
    a1[e] = (_Float16)xr[32 + k];
  }
  // B tile: W rows are output columns, contiguous K
  const float* wr = W + (size_t)(tn * 16 + lh) * 64;
  v16h b0, b1;
  #pragma unroll
  for (int e = 0; e < 16; ++e) {
    int k = half * 16 + e;
    b0[e] = (_Float16)wr[k];
    b1[e] = (_Float16)wr[32 + k];
  }
  v8f c = {};
  c = WMMA_F16(a0, b0, c);
  c = WMMA_F16(a1, b1, c);
  float bb = bias[tn * 16 + lh];
  #pragma unroll
  for (int r = 0; r < 8; ++r) c[r] += bb;

  if (job == 2) {                   // V stored transposed: VT[b][h][d][s]
    #pragma unroll
    for (int r = 0; r < 8; ++r) {
      int row = tm * 16 + r + 8 * half;          // global row in [0,4096)
      int bidx = row >> 11, s = row & 2047;
      int col = tn * 16 + lh;                    // global out col in [0,768)
      int hh = col >> 6, dd = col & 63;
      VT[(((size_t)(bidx * 12 + hh) * 64 + dd) << 11) + s] = (_Float16)c[r];
    }
  } else {
    _Float16* dst = (job == 0) ? Qh : (job == 1) ? Kh : (job == 3) ? pkP : pqP;
    #pragma unroll
    for (int r = 0; r < 8; ++r)
      dst[(size_t)(tm * 16 + r + 8 * half) * 768 + tn * 16 + lh] = (_Float16)c[r];
  }
}

// ---------------------------------------------------------------------------
// Fused flash attention with DeBERTa relative-position bias.
// One wave per (b, h, 16-query tile); loop over 64 key chunks of 32.
// Bias strips: positions [Delta-31, Delta+16] (48 wide, clamped to [0,1023]),
//   E[m][c] = q_{q0+m} . pk[clamp(Delta-31+c)]   (c2p)
//   F[n][c] = k_{k0+n} . pq[clamp(Delta-31+c)]   (p2c, idx = i-j+512 as well)
// score(m,n) += E[m][m-n+31] + F[n][m-n+31]
// ---------------------------------------------------------------------------
__global__ __launch_bounds__(32) void attn_kernel(
    const _Float16* __restrict__ Qh, const _Float16* __restrict__ Kh,
    const _Float16* __restrict__ VT, const _Float16* __restrict__ pkP,
    const _Float16* __restrict__ pqP, float* __restrict__ out)
{
  __shared__ float ldsE[16 * 48];
  __shared__ float ldsF[32 * 48];
  __shared__ float ldsP[16 * 32];

  const int lane = threadIdx.x, half = lane >> 4, lh = lane & 15;
  const int bid = blockIdx.x;
  const int qt = bid & 127, bh = bid >> 7;
  const int h = bh % 12, b = bh / 12;
  const int q0 = qt * 16;
  const float rscale = 0.07216878364870323f;   // 1/sqrt(64*3)

  // Q tile in A layout, K=64 split in two 16x32 halves
  v16h qa0, qa1;
  {
    const _Float16* qr = Qh + (size_t)(b * 2048 + q0 + lh) * 768 + h * 64;
    #pragma unroll
    for (int e = 0; e < 16; ++e) {
      int k = half * 16 + e;
      qa0[e] = qr[k];
      qa1[e] = qr[32 + k];
    }
  }

  float mrow[8], lrow[8];
  v8f acc[4];
  v8f zero = {};
  #pragma unroll
  for (int r = 0; r < 8; ++r) { mrow[r] = -3.0e38f; lrow[r] = 0.f; }
  #pragma unroll
  for (int dt = 0; dt < 4; ++dt) acc[dt] = zero;

  for (int k0 = 0; k0 < 2048; k0 += 32) {
    const int Dlt = q0 - k0 + 512;

    // ---- content scores q.k^T for 32 keys (two 16x16 tiles) ----
    v8f sc[2];
    v16h ka0[2], ka1[2];
    #pragma unroll
    for (int t = 0; t < 2; ++t) {
      const _Float16* kr = Kh + (size_t)(b * 2048 + k0 + t * 16 + lh) * 768 + h * 64;
      v16h kb0 = *(const v16h*)(kr + half * 16);        // B layout, K=d 0..31
      v16h kb1 = *(const v16h*)(kr + 32 + half * 16);   // K=d 32..63
      v8f c = zero;
      c = WMMA_F16(qa0, kb0, c);
      c = WMMA_F16(qa1, kb1, c);
      sc[t] = c;
      // same K rows in A layout for the p2c strip GEMM
      #pragma unroll
      for (int e = 0; e < 16; ++e) {
        int k = half * 16 + e;
        ka0[t][e] = kr[k];
        ka1[t][e] = kr[32 + k];
      }
    }

    // ---- positional strips E (c2p) and F (p2c), 48 positions ----
    #pragma unroll
    for (int tc = 0; tc < 3; ++tc) {
      int pos = Dlt - 31 + tc * 16 + lh;
      pos = pos < 0 ? 0 : (pos > 1023 ? 1023 : pos);
      const _Float16* pk = pkP + (size_t)pos * 768 + h * 64;
      v16h pb0 = *(const v16h*)(pk + half * 16);
      v16h pb1 = *(const v16h*)(pk + 32 + half * 16);
      v8f e8 = zero;
      e8 = WMMA_F16(qa0, pb0, e8);
      e8 = WMMA_F16(qa1, pb1, e8);
      #pragma unroll
      for (int r = 0; r < 8; ++r)
        ldsE[(r + 8 * half) * 48 + tc * 16 + lh] = e8[r];

      const _Float16* pq = pqP + (size_t)pos * 768 + h * 64;
      v16h qb0 = *(const v16h*)(pq + half * 16);
      v16h qb1 = *(const v16h*)(pq + 32 + half * 16);
      #pragma unroll
      for (int t = 0; t < 2; ++t) {
        v8f f8 = zero;
        f8 = WMMA_F16(ka0[t], qb0, f8);
        f8 = WMMA_F16(ka1[t], qb1, f8);
        #pragma unroll
        for (int r = 0; r < 8; ++r)
          ldsF[(t * 16 + r + 8 * half) * 48 + tc * 16 + lh] = f8[r];
      }
    }
    __syncthreads();

    // ---- bias add + online softmax (rows live in vgpr r, m = r+8*half) ----
    #pragma unroll
    for (int r = 0; r < 8; ++r) {
      int m = r + 8 * half;
      int n0 = lh, n1 = 16 + lh;
      float s0 = sc[0][r] + ldsE[m * 48 + (m - n0 + 31)] + ldsF[n0 * 48 + (m - n0 + 31)];
      float s1 = sc[1][r] + ldsE[m * 48 + (m - n1 + 31)] + ldsF[n1 * 48 + (m - n1 + 31)];
      s0 *= rscale; s1 *= rscale;
      float v = fmaxf(s0, s1);
      v = fmaxf(v, __shfl_xor(v, 8));
      v = fmaxf(v, __shfl_xor(v, 4));
      v = fmaxf(v, __shfl_xor(v, 2));
      v = fmaxf(v, __shfl_xor(v, 1));
      float mn   = fmaxf(mrow[r], v);
      float corr = __expf(mrow[r] - mn);
      float p0 = __expf(s0 - mn);
      float p1 = __expf(s1 - mn);
      float rs = p0 + p1;
      rs += __shfl_xor(rs, 8);
      rs += __shfl_xor(rs, 4);
      rs += __shfl_xor(rs, 2);
      rs += __shfl_xor(rs, 1);
      lrow[r] = lrow[r] * corr + rs;
      mrow[r] = mn;
      #pragma unroll
      for (int dt = 0; dt < 4; ++dt) acc[dt][r] *= corr;
      ldsP[m * 32 + n0] = p0;
      ldsP[m * 32 + n1] = p1;
    }
    __syncthreads();

    // ---- P in A layout (f16) then P @ V ----
    v16h pa;
    #pragma unroll
    for (int e = 0; e < 16; ++e) {
      int kkey = half * 16 + e;
      pa[e] = (_Float16)ldsP[lh * 32 + kkey];
    }
    const _Float16* vbase = VT + ((size_t)(b * 12 + h) << 17);  // *64*2048
    #pragma unroll
    for (int dt = 0; dt < 4; ++dt) {
      const _Float16* vr = vbase + ((size_t)(dt * 16 + lh) << 11) + k0 + half * 16;
      v16h vb = *(const v16h*)vr;                 // B layout: 16 contiguous keys
      acc[dt] = WMMA_F16(pa, vb, acc[dt]);
    }
    __syncthreads();
  }

  // ---- epilogue: O /= l, write f32 output [B,S,768] ----
  #pragma unroll
  for (int r = 0; r < 8; ++r) {
    int m = r + 8 * half;
    float inv = 1.0f / lrow[r];
    #pragma unroll
    for (int dt = 0; dt < 4; ++dt)
      out[(size_t)(b * 2048 + q0 + m) * 768 + h * 64 + dt * 16 + lh] = acc[dt][r] * inv;
  }
}

extern "C" void kernel_launch(void* const* d_in, const int* in_sizes, int n_in,
                              void* d_out, int out_size, void* d_ws, size_t ws_size,
                              hipStream_t stream) {
  const float* H   = (const float*)d_in[0];
  const float* P   = (const float*)d_in[1];
  // d_in[2] = relative_pos (recomputed analytically; unused)
  const float* Wq  = (const float*)d_in[3];  const float* bq  = (const float*)d_in[4];
  const float* Wk  = (const float*)d_in[5];  const float* bk  = (const float*)d_in[6];
  const float* Wv  = (const float*)d_in[7];  const float* bv  = (const float*)d_in[8];
  const float* Wpq = (const float*)d_in[9];  const float* bpq = (const float*)d_in[10];
  const float* Wpk = (const float*)d_in[11]; const float* bpk = (const float*)d_in[12];

  _Float16* Qh  = (_Float16*)d_ws;                 // [4096][768]
  _Float16* Kh  = Qh  + (size_t)4096 * 768;        // [4096][768]
  _Float16* VT  = Kh  + (size_t)4096 * 768;        // [2][12][64][2048]
  _Float16* pkP = VT  + (size_t)2 * 12 * 64 * 2048;// [1024][768]
  _Float16* pqP = pkP + (size_t)1024 * 768;        // [1024][768]

  proj_kernel<<<3 * 12288 + 2 * 3072, 32, 0, stream>>>(
      H, P, Wq, bq, Wk, bk, Wv, bv, Wpk, bpk, Wpq, bpq, Qh, Kh, VT, pkP, pqP);

  attn_kernel<<<2 * 12 * 128, 32, 0, stream>>>(Qh, Kh, VT, pkP, pqP, (float*)d_out);
}
